// LatentLanguage_23613730193610
// MI455X (gfx1250) — compile-verified, matrix operand order checked by reading, not run on hardware
//
#include <hip/hip_runtime.h>
#include <hip/hip_bf16.h>
#include <cmath>

// ---------------------------------------------------------------------------
// Model constants (reference file)
// ---------------------------------------------------------------------------
#define VOCAB 50257
#define DIM   768
#define STATE 16
#define CONVK 4
#define NB    2      // batch
#define NT    512    // seq len at outer stages

typedef __attribute__((ext_vector_type(16))) _Float16 v16h;
typedef __attribute__((ext_vector_type(8)))  _Float16 v8h;
typedef __attribute__((ext_vector_type(8)))  float    v8f;

// ===========================================================================
// Generic WMMA GEMM:  Y[M,N] = X[M,K] * W[N,K]^T (+ bias) (+ residual | softplus)
//   - block tile 128(M) x 64(N), 8 waves (4 M-groups x 2 N-groups)
//   - each wave: 32x32 via 2 A-frags x 2 B-frags -> 4 WMMA / K-step
//   - fast path (interior block, K%32==0): unguarded float4 loads,
//     register-pipelined so next tile's global loads overlap the WMMAs
//   mode: 0 = bias only, 1 = bias + residual add, 2 = bias + softplus
// ===========================================================================
#define GTM 128
#define GTN 64
#define GTK 32

__device__ __forceinline__ v8h cvt8(float4 p0, float4 p1)
{
    v8h h;
    h[0] = (_Float16)p0.x; h[1] = (_Float16)p0.y;
    h[2] = (_Float16)p0.z; h[3] = (_Float16)p0.w;
    h[4] = (_Float16)p1.x; h[5] = (_Float16)p1.y;
    h[6] = (_Float16)p1.z; h[7] = (_Float16)p1.w;
    return h;
}

__global__ __launch_bounds__(256)
void gemm_wmma_kernel(const float* __restrict__ X, int ldx,
                      const float* __restrict__ W,            // (N,K) row-major
                      const float* __restrict__ bias,
                      const float* __restrict__ residual, int ldres,
                      float* __restrict__ Y, int ldy,
                      int M, int N, int K, int mode)
{
    __shared__ _Float16 As[GTM][GTK];   // 128x32 f16 = 8KB
    __shared__ _Float16 Bs[GTN][GTK];   //  64x32 f16 = 4KB

    const int tid  = threadIdx.x;
    const int lane = tid & 31;
    const int wave = tid >> 5;          // 0..7
    const int wm   = wave >> 1;         // 0..3 -> 32-row group
    const int wn   = wave & 1;          // 0..1 -> 32-col group
    const int blockM = blockIdx.y * GTM;
    const int blockN = blockIdx.x * GTN;

    // staging coords: each thread fills 8 contiguous K elements per row-group
    const int sr = tid >> 2;            // 0..63
    const int sc = (tid & 3) << 3;      // 0,8,16,24

    v8f acc00 = {}, acc01 = {}, acc10 = {}, acc11 = {};

    // fragment compute from LDS (ISA 7.12.2 16-bit A 16x32 layout):
    // lane L: row = L%16; halves 0-7 -> K khi..khi+7, halves 8-15 -> K khi+16..khi+23
    const int frow = lane & 15;
    const int khi  = (lane >> 4) << 3;
    auto loadfrag = [&](const _Float16* base) -> v16h {
        v8h lo = *(const v8h*)(base + khi);
        v8h hi = *(const v8h*)(base + khi + 16);
        return __builtin_shufflevector(lo, hi, 0,1,2,3,4,5,6,7,8,9,10,11,12,13,14,15);
    };
    auto compute = [&]() {
        v16h a0 = loadfrag(&As[(wm << 5) + frow][0]);
        v16h a1 = loadfrag(&As[(wm << 5) + 16 + frow][0]);
        v16h b0 = loadfrag(&Bs[(wn << 5) + frow][0]);
        v16h b1 = loadfrag(&Bs[(wn << 5) + 16 + frow][0]);
        acc00 = __builtin_amdgcn_wmma_f32_16x16x32_f16(false, a0, false, b0, (short)0, acc00, false, false);
        acc01 = __builtin_amdgcn_wmma_f32_16x16x32_f16(false, a0, false, b1, (short)0, acc01, false, false);
        acc10 = __builtin_amdgcn_wmma_f32_16x16x32_f16(false, a1, false, b0, (short)0, acc10, false, false);
        acc11 = __builtin_amdgcn_wmma_f32_16x16x32_f16(false, a1, false, b1, (short)0, acc11, false, false);
    };

    const bool fast = (blockM + GTM <= M) && (blockN + GTN <= N) && ((K & (GTK - 1)) == 0);

    if (fast) {
        // ---- unguarded, register-pipelined path (dominant case) ----
        const float* pA0 = X + (size_t)(blockM + sr) * ldx + sc;
        const float* pA1 = X + (size_t)(blockM + 64 + sr) * ldx + sc;
        const float* pB  = W + (size_t)(blockN + sr) * K + sc;

        float4 ra0 = *(const float4*)pA0, ra1 = *(const float4*)(pA0 + 4);
        float4 ra2 = *(const float4*)pA1, ra3 = *(const float4*)(pA1 + 4);
        float4 rb0 = *(const float4*)pB,  rb1 = *(const float4*)(pB + 4);

        for (int k0 = 0; k0 < K; k0 += GTK) {
            *(v8h*)&As[sr][sc]      = cvt8(ra0, ra1);
            *(v8h*)&As[sr + 64][sc] = cvt8(ra2, ra3);
            *(v8h*)&Bs[sr][sc]      = cvt8(rb0, rb1);
            __syncthreads();
            if (k0 + GTK < K) {
                pA0 += GTK; pA1 += GTK; pB += GTK;
                ra0 = *(const float4*)pA0; ra1 = *(const float4*)(pA0 + 4);
                ra2 = *(const float4*)pA1; ra3 = *(const float4*)(pA1 + 4);
                rb0 = *(const float4*)pB;  rb1 = *(const float4*)(pB + 4);
                if (k0 + 2 * GTK < K)
                    __builtin_prefetch(pB + GTK, 0, 1);   // global_prefetch_b8
            }
            compute();
            __syncthreads();
        }
    } else {
        // ---- fully guarded path (edge blocks / ragged K) ----
        for (int k0 = 0; k0 < K; k0 += GTK) {
            {
                int gm = blockM + sr, gk = k0 + sc;
                v8h h;
                #pragma unroll
                for (int j = 0; j < 8; ++j)
                    h[j] = (_Float16)((gm < M && gk + j < K) ? X[(size_t)gm * ldx + gk + j] : 0.0f);
                *(v8h*)&As[sr][sc] = h;
            }
            {
                int gm = blockM + 64 + sr, gk = k0 + sc;
                v8h h;
                #pragma unroll
                for (int j = 0; j < 8; ++j)
                    h[j] = (_Float16)((gm < M && gk + j < K) ? X[(size_t)gm * ldx + gk + j] : 0.0f);
                *(v8h*)&As[sr + 64][sc] = h;
            }
            {
                int gn = blockN + sr, gk = k0 + sc;
                v8h h;
                #pragma unroll
                for (int j = 0; j < 8; ++j)
                    h[j] = (_Float16)((gn < N && gk + j < K) ? W[(size_t)gn * K + gk + j] : 0.0f);
                *(v8h*)&Bs[sr][sc] = h;
            }
            __syncthreads();
            compute();
            __syncthreads();
        }
    }

    // ---- store: C/D layout: lanes 0-15 -> N=lane, M=vgpr; lanes 16-31 -> M=vgpr+8
    const int mB = blockM + (wm << 5) + ((lane >> 4) << 3);
    const int nB = blockN + (wn << 5) + (lane & 15);
    #pragma unroll
    for (int i = 0; i < 2; ++i) {
        #pragma unroll
        for (int j = 0; j < 2; ++j) {
            v8f acc = (i == 0) ? ((j == 0) ? acc00 : acc01)
                               : ((j == 0) ? acc10 : acc11);
            const int n = nB + (j << 4);
            if (n < N) {
                const float bv = bias ? bias[n] : 0.0f;
                #pragma unroll
                for (int v = 0; v < 8; ++v) {
                    int m = mB + (i << 4) + v;
                    if (m < M) {
                        float r = acc[v] + bv;
                        if (mode == 1) {
                            r += residual[(size_t)m * ldres + n];
                        } else if (mode == 2) {       // softplus
                            r = (r > 20.0f) ? r : log1pf(__expf(r));
                        }
                        Y[(size_t)m * ldy + n] = r;
                    }
                }
            }
        }
    }
}

// ===========================================================================
// Elementwise / per-row kernels
// ===========================================================================
__global__ void embed_kernel(const int* __restrict__ tok,
                             const float* __restrict__ emb,
                             float* __restrict__ x, int dim)
{
    int t = blockIdx.x;
    const float* src = emb + (size_t)tok[t] * dim;
    float* dst = x + (size_t)t * dim;
    for (int d = threadIdx.x; d < dim; d += blockDim.x) dst[d] = src[d];
}

__global__ void rmsnorm_kernel(const float* __restrict__ x,
                               const float* __restrict__ w,
                               float* __restrict__ y, int dim)
{
    __shared__ float red[8];
    const int row = blockIdx.x;
    const float* xr = x + (size_t)row * dim;
    float s = 0.0f;
    for (int d = threadIdx.x; d < dim; d += blockDim.x) { float v = xr[d]; s += v * v; }
    #pragma unroll
    for (int off = 16; off > 0; off >>= 1) s += __shfl_down(s, off, 32);
    int lane = threadIdx.x & 31, wv = threadIdx.x >> 5;
    if (lane == 0) red[wv] = s;
    __syncthreads();
    if (threadIdx.x == 0) {
        float t = 0.0f;
        for (int i = 0; i < (int)(blockDim.x >> 5); ++i) t += red[i];
        red[0] = rsqrtf(t / (float)dim + 1e-5f);
    }
    __syncthreads();
    const float scale = red[0];
    for (int d = threadIdx.x; d < dim; d += blockDim.x)
        y[(size_t)row * dim + d] = xr[d] * scale * w[d];
}

__global__ void split_kernel(const float* __restrict__ xz,
                             float* __restrict__ xin, float* __restrict__ z,
                             size_t total, int din)
{
    size_t i = (size_t)blockIdx.x * blockDim.x + threadIdx.x;
    if (i >= total) return;
    size_t m = i / din; int d = (int)(i % din);
    xin[i] = xz[m * (size_t)(2 * din) + d];
    z[i]   = xz[m * (size_t)(2 * din) + din + d];
}

__global__ void conv_silu_kernel(const float* __restrict__ x,
                                 const float* __restrict__ cw,  // (din,4)
                                 const float* __restrict__ cb,
                                 float* __restrict__ out,
                                 int T, int din, size_t total)
{
    size_t i = (size_t)blockIdx.x * blockDim.x + threadIdx.x;
    if (i >= total) return;
    int d = (int)(i % din);
    size_t bt = i / din;
    int t = (int)(bt % T);
    int b = (int)(bt / T);
    float acc = cb[d];
    #pragma unroll
    for (int j = 0; j < CONVK; ++j) {
        int tt = t - (CONVK - 1) + j;
        if (tt >= 0) acc += x[((size_t)b * T + tt) * din + d] * cw[d * CONVK + j];
    }
    out[i] = acc / (1.0f + __expf(-acc));   // silu
}

__global__ void scan_kernel(const float* __restrict__ u,
                            const float* __restrict__ delta,
                            const float* __restrict__ A_log,   // (din,16)
                            const float* __restrict__ Dp,      // (din,)
                            const float* __restrict__ xdbc,    // (B*T, ldxd); Bm at +dtr, Cm at +dtr+16
                            int ldxd, int dtr,
                            float* __restrict__ y,
                            int T, int din)
{
    int d = blockIdx.x * blockDim.x + threadIdx.x;
    int b = blockIdx.y;
    if (d >= din) return;

    float A[STATE];
    #pragma unroll
    for (int s = 0; s < STATE; ++s) A[s] = -__expf(A_log[(size_t)d * STATE + s]);
    float h[STATE];
    #pragma unroll
    for (int s = 0; s < STATE; ++s) h[s] = 0.0f;
    const float Dd = Dp[d];

    for (int t = 0; t < T; ++t) {
        size_t bt = (size_t)b * T + t;
        float dl = delta[bt * din + d];
        float uu = u[bt * din + d];
        const float* Bm = xdbc + bt * ldxd + dtr;
        const float* Cm = Bm + STATE;
        float acc = 0.0f;
        #pragma unroll
        for (int s = 0; s < STATE; ++s) {
            h[s] = __expf(dl * A[s]) * h[s] + dl * Bm[s] * uu;
            acc += h[s] * Cm[s];
        }
        y[bt * din + d] = acc + uu * Dd;
    }
}

__global__ void gate_kernel(float* __restrict__ y, const float* __restrict__ z, size_t n)
{
    size_t i = (size_t)blockIdx.x * blockDim.x + threadIdx.x;
    if (i >= n) return;
    float zv = z[i];
    y[i] *= zv / (1.0f + __expf(-zv));
}

__global__ void ngram_concat_kernel(const float* __restrict__ x,
                                    float* __restrict__ cat,
                                    int n, int T, int dim, size_t total)
{
    size_t i = (size_t)blockIdx.x * blockDim.x + threadIdx.x;
    if (i >= total) return;
    int d = (int)(i % dim);
    size_t bt = i / dim;
    int t = (int)(bt % T);
    int b = (int)(bt / T);
    cat[bt * (size_t)(2 * dim) + d] = x[i];
    int ts = t - n;
    cat[bt * (size_t)(2 * dim) + dim + d] =
        (ts >= 0) ? x[((size_t)b * T + ts) * dim + d] : 0.0f;
}

// ===========================================================================
// Host-side orchestration
// ===========================================================================
namespace {

struct Lin   { const float* b; const float* w; };
struct MambaP {
    const float* A_log; const float* D;
    const float* conv_b; const float* conv_w;
    Lin dt; Lin in;
    const float* norm; Lin out;
    const float* x_proj;
    int dim;
};

struct Scratch {
    float *xa, *xb, *xn, *cat, *xz, *xin, *z, *xc, *xdbc, *delta, *y;
};

inline void launch_gemm(hipStream_t s,
                        const float* X, int ldx,
                        const float* W, const float* bias,
                        const float* res, int ldres,
                        float* Y, int ldy,
                        int M, int N, int K, int mode)
{
    dim3 grid((N + GTN - 1) / GTN, (M + GTM - 1) / GTM);
    gemm_wmma_kernel<<<grid, dim3(256), 0, s>>>(X, ldx, W, bias, res, ldres,
                                                Y, ldy, M, N, K, mode);
}

inline void run_mamba(hipStream_t s, const MambaP& m, float* x,
                      int Bn, int T, const Scratch& w)
{
    const int dim = m.dim;
    const int din = 2 * dim;                 // EXPAND * dim
    const int dtr = (dim + 15) / 16;
    const int nxd = dtr + 2 * STATE;
    const int M   = Bn * T;
    const size_t tot = (size_t)M * din;
    const int TPB = 256;

    // xn = rmsnorm(x)
    rmsnorm_kernel<<<M, TPB, 0, s>>>(x, m.norm, w.xn, dim);
    // xz = xn @ in_w^T + in_b            (N = 2*din)
    launch_gemm(s, w.xn, dim, m.in.w, m.in.b, nullptr, 0, w.xz, 2 * din,
                M, 2 * din, dim, 0);
    // split into xin / z
    split_kernel<<<(unsigned)((tot + TPB - 1) / TPB), TPB, 0, s>>>(w.xz, w.xin, w.z, tot, din);
    // xc = silu(causal depthwise conv(xin))
    conv_silu_kernel<<<(unsigned)((tot + TPB - 1) / TPB), TPB, 0, s>>>(
        w.xin, m.conv_w, m.conv_b, w.xc, T, din, tot);
    // xdbc = xc @ x_proj^T   (no bias)
    launch_gemm(s, w.xc, din, m.x_proj, nullptr, nullptr, 0, w.xdbc, nxd,
                M, nxd, din, 0);
    // delta = softplus(dt @ dt_w^T + dt_b)   (K = dtr, X strided in xdbc)
    launch_gemm(s, w.xdbc, nxd, m.dt.w, m.dt.b, nullptr, 0, w.delta, din,
                M, din, dtr, 2);
    // selective scan
    dim3 sg((din + TPB - 1) / TPB, Bn);
    scan_kernel<<<sg, TPB, 0, s>>>(w.xc, w.delta, m.A_log, m.D,
                                   w.xdbc, nxd, dtr, w.y, T, din);
    // y *= silu(z)
    gate_kernel<<<(unsigned)((tot + TPB - 1) / TPB), TPB, 0, s>>>(w.y, w.z, tot);
    // x = x + y @ out_w^T + out_b   (in-place residual)
    launch_gemm(s, w.y, din, m.out.w, m.out.b, x, dim, x, dim,
                M, dim, din, 1);
}

} // namespace

extern "C" void kernel_launch(void* const* d_in, const int* in_sizes, int n_in,
                              void* d_out, int out_size, void* d_ws, size_t ws_size,
                              hipStream_t stream)
{
    (void)in_sizes; (void)out_size; (void)ws_size;

    // ---- decode params: recursive sorted-key (pytree) flatten order ----
    int idx = 0;
    auto F = [&]() { return (const float*)d_in[idx++]; };
    auto readLin = [&]() { Lin l; l.b = F(); l.w = F(); return l; };
    auto readM = [&](int dim) {
        MambaP m;
        m.A_log = F(); m.D = F();
        m.conv_b = F(); m.conv_w = F();
        m.dt = readLin(); m.in = readLin();
        m.norm = F(); m.out = readLin();
        m.x_proj = F();
        m.dim = dim;
        return m;
    };

    MambaP body[4][2];
    for (int b = 0; b < 4; ++b)
        for (int j = 0; j < 2; ++j) body[b][j] = readM(DIM / 2);
    MambaP md0 = readM(DIM / 2);
    MambaP md1 = readM(DIM);
    MambaP md2 = readM(DIM);
    MambaP md3 = readM(DIM);
    Lin dec_lin = readLin();
    const float* emb = F();
    Lin enc_lin = readLin();
    Lin ex1 = readLin();
    Lin ex2 = readLin();
    const float* lm_b = F();
    MambaP m0 = readM(DIM);
    MambaP m1 = readM(DIM);
    MambaP m2 = readM(DIM);
    MambaP m3 = readM(DIM);
    MambaP m4 = readM(DIM / 2);
    Lin mg1 = readLin();
    Lin mg2 = readLin();
    Lin ng[3];
    for (int i = 0; i < 3; ++i) ng[i] = readLin();
    const int* tokens = (const int*)d_in[n_in - 1];

    // ---- carve workspace (all sizes in floats) ----
    const size_t S_X    = (size_t)NB * NT * DIM;          //  786432
    const size_t S_CAT  = (size_t)NB * NT * 2 * DIM;      // 1572864
    const size_t S_XZ   = (size_t)NB * NT * 4 * DIM;      // 3145728
    const size_t S_DIN  = (size_t)NB * NT * 2 * DIM;      // 1572864
    const size_t S_XDBC = (size_t)NB * NT * 80;           //   81920

    float* p = (float*)d_ws;
    Scratch ws;
    ws.xa    = p; p += S_X;
    ws.xb    = p; p += S_X;
    ws.xn    = p; p += S_X;
    ws.cat   = p; p += S_CAT;
    ws.xz    = p; p += S_XZ;
    ws.xin   = p; p += S_DIN;
    ws.z     = p; p += S_DIN;
    ws.xc    = p; p += S_DIN;
    ws.xdbc  = p; p += S_XDBC;
    ws.delta = p; p += S_DIN;
    ws.y     = p; p += S_DIN;

    float* cur = ws.xa;
    float* alt = ws.xb;
    auto swapx = [&]() { float* t = cur; cur = alt; alt = t; };

    // ---- forward pass ----
    embed_kernel<<<NB * NT, 256, 0, stream>>>(tokens, emb, cur, DIM);

    run_mamba(stream, m0, cur, NB, NT, ws);

    // ngram layers (residual, in-place)
    for (int i = 0; i < 3; ++i) {
        size_t tot = (size_t)NB * NT * DIM;
        ngram_concat_kernel<<<(unsigned)((tot + 255) / 256), 256, 0, stream>>>(
            cur, ws.cat, i + 1, NT, DIM, tot);
        launch_gemm(stream, ws.cat, 2 * DIM, ng[i].w, ng[i].b, cur, DIM,
                    cur, DIM, NB * NT, DIM, 2 * DIM, 1);
    }

    run_mamba(stream, m1, cur, NB, NT, ws);

    // merge 1: (B,512,768) -> (B,256,768)
    launch_gemm(stream, cur, 2 * DIM, mg1.w, mg1.b, nullptr, 0, alt, DIM,
                NB * (NT / 2), DIM, 2 * DIM, 0);
    swapx();
    run_mamba(stream, m2, cur, NB, NT / 2, ws);

    // merge 2: -> (B,128,768)
    launch_gemm(stream, cur, 2 * DIM, mg2.w, mg2.b, nullptr, 0, alt, DIM,
                NB * (NT / 4), DIM, 2 * DIM, 0);
    swapx();
    run_mamba(stream, m3, cur, NB, NT / 4, ws);

    // enc_lin: 768 -> 384
    launch_gemm(stream, cur, DIM, enc_lin.w, enc_lin.b, nullptr, 0, alt, DIM / 2,
                NB * (NT / 4), DIM / 2, DIM, 0);
    swapx();
    run_mamba(stream, m4, cur, NB, NT / 4, ws);

    // body: 4 x 2 mamba blocks @ dim 384
    for (int b = 0; b < 4; ++b) {
        run_mamba(stream, body[b][0], cur, NB, NT / 4, ws);
        run_mamba(stream, body[b][1], cur, NB, NT / 4, ws);
    }
    run_mamba(stream, md0, cur, NB, NT / 4, ws);

    // dec_lin: 384 -> 768
    launch_gemm(stream, cur, DIM / 2, dec_lin.w, dec_lin.b, nullptr, 0, alt, DIM,
                NB * (NT / 4), DIM, DIM / 2, 0);
    swapx();
    run_mamba(stream, md1, cur, NB, NT / 4, ws);

    // expand 1: (B,128,768) -> (B,256,768)
    launch_gemm(stream, cur, DIM, ex1.w, ex1.b, nullptr, 0, alt, 2 * DIM,
                NB * (NT / 4), 2 * DIM, DIM, 0);
    swapx();
    run_mamba(stream, md2, cur, NB, NT / 2, ws);

    // expand 2: -> (B,512,768)
    launch_gemm(stream, cur, DIM, ex2.w, ex2.b, nullptr, 0, alt, 2 * DIM,
                NB * (NT / 2), 2 * DIM, DIM, 0);
    swapx();
    run_mamba(stream, md3, cur, NB, NT, ws);

    // LM head: (B*T, 768) @ emb(50257,768)^T + lm_b -> d_out
    launch_gemm(stream, cur, DIM, emb, lm_b, nullptr, 0,
                (float*)d_out, VOCAB, NB * NT, VOCAB, DIM, 0);
}
